// Attention_77541339562539
// MI455X (gfx1250) — matrix operand order repaired and checked by reading, not compiled, not run on hardware
//
#include <hip/hip_runtime.h>

// ---------------------------------------------------------------------------
// CDNA5 (gfx1250) implementation: grid-window attention + pooled global
// attention block. All matmuls via v_wmma_f32_16x16x32_bf16; GEMM tiles are
// staged with GLOBAL_LOAD_ASYNC_TO_LDS (ASYNCcnt) and double-buffered LDS.
// ---------------------------------------------------------------------------

typedef __attribute__((ext_vector_type(16))) __bf16 v16bf;
typedef __attribute__((ext_vector_type(8)))  float  v8f;
typedef __attribute__((ext_vector_type(4)))  int    v4i_;

#define WMMA_BF16(A, B, C) \
  __builtin_amdgcn_wmma_f32_16x16x32_bf16(false, (A), false, (B), (short)0, (C), false, false)

__device__ __forceinline__ float bf2f(unsigned short u) {
  unsigned int x = ((unsigned int)u) << 16;
  float f;
  __builtin_memcpy(&f, &x, 4);
  return f;
}
__device__ __forceinline__ unsigned short f2bf(float f) {
  unsigned int x;
  __builtin_memcpy(&x, &f, 4);
  unsigned int r = x + 0x7FFFu + ((x >> 16) & 1u);   // round-to-nearest-even
  return (unsigned short)(r >> 16);
}
__device__ __forceinline__ __bf16 us2bf(unsigned short u) {
  return __builtin_bit_cast(__bf16, u);
}

// --- async global -> LDS copy (16 bytes per lane), ASYNCcnt tracked --------
__device__ __forceinline__ void async_copy_b128(unsigned short* lds, const unsigned short* g) {
#if __has_builtin(__builtin_amdgcn_global_load_async_to_lds_b128)
  __builtin_amdgcn_global_load_async_to_lds_b128(
      (__attribute__((address_space(1))) v4i_*)(unsigned long long)g,
      (__attribute__((address_space(3))) v4i_*)(unsigned)(unsigned long long)lds,
      0, 0);
#else
  unsigned loff = (unsigned)(unsigned long long)lds;   // generic LDS ptr: low 32 bits = LDS offset
  asm volatile("global_load_async_to_lds_b128 %0, %1, off" :: "v"(loff), "v"(g) : "memory");
#endif
}

#if __has_builtin(__builtin_amdgcn_s_wait_asynccnt)
#define WAIT_ASYNC(n) __builtin_amdgcn_s_wait_asynccnt(n)
#else
#define WAIT_ASYNC(n) asm volatile("s_wait_asynccnt %0" :: "n"(n) : "memory")
#endif

// ---------------------------------------------------------------------------
// fp32 -> bf16 conversion (inputs + weights)
// ---------------------------------------------------------------------------
__global__ __launch_bounds__(256) void f32_to_bf16_kernel(const float* __restrict__ in,
                                                          unsigned short* __restrict__ out,
                                                          int n) {
  int i = blockIdx.x * 256 + threadIdx.x;
  if (i < n) out[i] = f2bf(in[i]);
}

// ---------------------------------------------------------------------------
// GEMM: out[m,o] = sum_k A[m,k] * W[o,k] + bias[o]
// A: M x K bf16 row-major, W: O x K bf16 row-major (conv1x1 weight layout).
// Block tile 128x64, 8 waves, each wave a 32x32 patch (4 WMMA accumulators).
// Tiles streamed via async-to-LDS with double buffering.
// ---------------------------------------------------------------------------
__global__ __launch_bounds__(256) void gemm_bias_kernel(const unsigned short* __restrict__ A,
                                                        const unsigned short* __restrict__ W,
                                                        const float* __restrict__ bias,
                                                        unsigned short* __restrict__ outb,
                                                        float* __restrict__ outf,
                                                        int M, int K, int O) {
  __shared__ unsigned short As[2][128][40];  // padded rows (80B stride, 16B aligned)
  __shared__ unsigned short Bs[2][64][40];

  const int tid  = threadIdx.x;
  const int lane = tid & 31;
  const int wv   = tid >> 5;      // 0..7
  const int wm   = wv & 3;        // 32-row strip within block
  const int wo   = wv >> 2;       // 0..1 -> 32-col half of O tile
  const int m0   = blockIdx.x * 128;
  const int o0   = blockIdx.y * 64;
  const int r    = lane & 15;
  const int hi   = lane >> 4;

  const int lrow = tid >> 2;       // 0..63
  const int lcol = (tid & 3) * 8;  // 0,8,16,24 (bf16 elements; 16B chunks)

  v8f acc00 = {}, acc01 = {}, acc10 = {}, acc11 = {};

  auto issue_tile = [&](int k0, int buf) {
    const unsigned short* ap0 = A + (size_t)(m0 + lrow) * K + k0 + lcol;
    const unsigned short* ap1 = A + (size_t)(m0 + 64 + lrow) * K + k0 + lcol;
    const unsigned short* wp  = W + (size_t)(o0 + lrow) * K + k0 + lcol;
    async_copy_b128(&As[buf][lrow][lcol], ap0);
    async_copy_b128(&As[buf][64 + lrow][lcol], ap1);
    async_copy_b128(&Bs[buf][lrow][lcol], wp);
  };

  const int nk = K / 32;
  issue_tile(0, 0);

  for (int kk = 0; kk < nk; ++kk) {
    const int buf = kk & 1;
    if (kk + 1 < nk) {
      issue_tile((kk + 1) * 32, buf ^ 1);   // prefetch next slice into other buffer
      WAIT_ASYNC(3);                        // in-order completion: current tile landed
    } else {
      WAIT_ASYNC(0);
    }
    __syncthreads();

    v16bf a0, a1, b0, b1;
#pragma unroll
    for (int i = 0; i < 8; ++i) {
      a0[i]     = us2bf(As[buf][wm * 32 + r][hi * 8 + i]);
      a0[i + 8] = us2bf(As[buf][wm * 32 + r][16 + hi * 8 + i]);
      a1[i]     = us2bf(As[buf][wm * 32 + 16 + r][hi * 8 + i]);
      a1[i + 8] = us2bf(As[buf][wm * 32 + 16 + r][16 + hi * 8 + i]);
    }
#pragma unroll
    for (int i = 0; i < 16; ++i) {
      b0[i] = us2bf(Bs[buf][wo * 32 + r][(i < 8 ? hi * 8 + i : 8 + hi * 8 + i)]);
      b1[i] = us2bf(Bs[buf][wo * 32 + 16 + r][(i < 8 ? hi * 8 + i : 8 + hi * 8 + i)]);
    }
    acc00 = WMMA_BF16(a0, b0, acc00);
    acc01 = WMMA_BF16(a0, b1, acc01);
    acc10 = WMMA_BF16(a1, b0, acc10);
    acc11 = WMMA_BF16(a1, b1, acc11);
    __syncthreads();
  }

  auto store_tile = [&](const v8f& acc, int mt, int ot) {
#pragma unroll
    for (int j = 0; j < 8; ++j) {
      int rowo = m0 + wm * 32 + mt * 16 + j + 8 * hi;  // D layout: M = j + 8*(lane>=16)
      int c    = o0 + wo * 32 + ot * 16 + r;
      float v  = acc[j] + bias[c];
      if (outf) outf[(size_t)rowo * O + c] = v;
      else      outb[(size_t)rowo * O + c] = f2bf(v);
    }
  };
  store_tile(acc00, 0, 0);
  store_tile(acc01, 0, 1);
  store_tile(acc10, 1, 0);
  store_tile(acc11, 1, 1);
}

// ---------------------------------------------------------------------------
// Grid-window attention: one wave per (b, head, hh, ww) window of 16 tokens.
// Writes z = x + grid_attn (fp32) for the following channel LayerNorm.
// ---------------------------------------------------------------------------
__global__ __launch_bounds__(128) void grid_attn_kernel(const unsigned short* __restrict__ qkvb,  // 65536 x 768
                                                        const float* __restrict__ x,              // (B,N,C)
                                                        float* __restrict__ zbuf) {               // (B,N,C)
  __shared__ float Pb[4][16][17];
  const int lane = threadIdx.x & 31;
  const int wv   = threadIdx.x >> 5;
  const int win  = blockIdx.x * 4 + wv;       // 16384 windows
  const int ww   = win & 15;
  const int hh   = (win >> 4) & 15;
  const int head = (win >> 8) & 3;
  const int b    = win >> 10;
  const int r    = lane & 15;
  const int hi   = lane >> 4;
  const float scale = 0.125f;                 // 1/sqrt(64)

  auto tok = [&](int t) { return b * 4096 + (hh * 4 + (t >> 2)) * 64 + (ww * 4 + (t & 3)); };

  // Q as A-operand (rows = query tokens)
  const unsigned short* qrow = qkvb + (size_t)tok(r) * 768 + head * 64;
  v16bf a0, a1;
#pragma unroll
  for (int i = 0; i < 8; ++i) {
    a0[i]     = us2bf(qrow[hi * 8 + i]);
    a0[i + 8] = us2bf(qrow[16 + hi * 8 + i]);
    a1[i]     = us2bf(qrow[32 + hi * 8 + i]);
    a1[i + 8] = us2bf(qrow[48 + hi * 8 + i]);
  }
  // K^T as B-operand (columns = key tokens)
  const unsigned short* krow = qkvb + (size_t)tok(r) * 768 + 256 + head * 64;
  v16bf b0, b1;
#pragma unroll
  for (int i = 0; i < 16; ++i) {
    b0[i] = us2bf(krow[(i < 8 ? hi * 8 + i : 8 + hi * 8 + i)]);
    b1[i] = us2bf(krow[32 + (i < 8 ? hi * 8 + i : 8 + hi * 8 + i)]);
  }
  v8f s = {};
  s = WMMA_BF16(a0, b0, s);
  s = WMMA_BF16(a1, b1, s);

  // softmax over key dim (= lanes within 16-group of D layout)
  float p[8];
#pragma unroll
  for (int j = 0; j < 8; ++j) {
    float v  = s[j] * scale;
    float mx = v;
#pragma unroll
    for (int d = 8; d >= 1; d >>= 1) mx = fmaxf(mx, __shfl_xor(mx, d, 32));
    float e   = __expf(v - mx);
    float sum = e;
#pragma unroll
    for (int d = 8; d >= 1; d >>= 1) sum += __shfl_xor(sum, d, 32);
    p[j] = e / sum;
  }

  // P (D layout) -> LDS -> A operand, K padded 16->32 with zeros
#pragma unroll
  for (int j = 0; j < 8; ++j) Pb[wv][j + 8 * hi][r] = p[j];
  __syncthreads();
  v16bf pa;
#pragma unroll
  for (int i = 0; i < 8; ++i) {
    pa[i]     = us2bf(f2bf(Pb[wv][r][hi * 8 + i]));
    pa[i + 8] = us2bf((unsigned short)0);
  }
  __syncthreads();

  // out = P @ V (four 16-dim chunks), write z = x + out
#pragma unroll
  for (int nd = 0; nd < 4; ++nd) {
    v16bf vb;
#pragma unroll
    for (int i = 0; i < 16; ++i) {
      int k = (i < 8 ? hi * 8 + i : 8 + hi * 8 + i);
      unsigned short u = 0;
      if (k < 16) u = qkvb[(size_t)tok(k) * 768 + 512 + head * 64 + nd * 16 + r];
      vb[i] = us2bf(u);
    }
    v8f o = {};
    o = WMMA_BF16(pa, vb, o);
#pragma unroll
    for (int j = 0; j < 8; ++j) {
      int m  = tok(j + 8 * hi);
      int ch = head * 64 + nd * 16 + r;
      zbuf[(size_t)m * 256 + ch] = x[(size_t)m * 256 + ch] + o[j];
    }
  }
}

// ---------------------------------------------------------------------------
// Channels-first LayerNorm over C=256 per token, output bf16.
// ---------------------------------------------------------------------------
__global__ __launch_bounds__(256) void layernorm_kernel(const float* __restrict__ zbuf,
                                                        const float* __restrict__ gw,
                                                        const float* __restrict__ gb,
                                                        unsigned short* __restrict__ lnb) {
  const int lane = threadIdx.x & 31;
  const int wv   = threadIdx.x >> 5;
  const int m    = blockIdx.x * 8 + wv;
  const float* z = zbuf + (size_t)m * 256;

  float vals[8];
  float s = 0.f;
#pragma unroll
  for (int i = 0; i < 8; ++i) {
    vals[i] = z[i * 32 + lane];
    s += vals[i];
  }
#pragma unroll
  for (int d = 16; d >= 1; d >>= 1) s += __shfl_xor(s, d, 32);
  float u  = s * (1.f / 256.f);
  float vs = 0.f;
#pragma unroll
  for (int i = 0; i < 8; ++i) {
    float t = vals[i] - u;
    vs += t * t;
  }
#pragma unroll
  for (int d = 16; d >= 1; d >>= 1) vs += __shfl_xor(vs, d, 32);
  float rinv = rsqrtf(vs * (1.f / 256.f) + 1e-6f);
#pragma unroll
  for (int i = 0; i < 8; ++i) {
    int c = i * 32 + lane;
    lnb[(size_t)m * 256 + c] = f2bf(gw[c] * (vals[i] - u) * rinv + gb[c]);
  }
}

// ---------------------------------------------------------------------------
// 4x4 average pooling of LN output: (B,C,64,64) -> (B*256, 256) bf16 rows.
// ---------------------------------------------------------------------------
__global__ __launch_bounds__(256) void pool_kernel(const unsigned short* __restrict__ lnb,
                                                   unsigned short* __restrict__ pooledb) {
  int idx = blockIdx.x * 256 + threadIdx.x;   // B*256*256
  int c   = idx & 255;
  int pp  = (idx >> 8) & 255;
  int b   = idx >> 16;
  int ph = pp >> 4, pw = pp & 15;
  float s = 0.f;
#pragma unroll
  for (int i = 0; i < 4; ++i)
#pragma unroll
    for (int j = 0; j < 4; ++j) {
      int m = b * 4096 + (ph * 4 + i) * 64 + (pw * 4 + j);
      s += bf2f(lnb[(size_t)m * 256 + c]);
    }
  pooledb[(size_t)idx] = f2bf(s * (1.f / 16.f));
}

// ---------------------------------------------------------------------------
// Pooled global attention with flash-style online softmax.
// One wave = 16 query rows of one (b, head); 256 keys in 16 chunks.
// Writes gsum = attn_out + ln (bf16) for the final projection.
// ---------------------------------------------------------------------------
__global__ __launch_bounds__(256) void global_attn_kernel(const unsigned short* __restrict__ qgb,   // 65536 x 256
                                                          const unsigned short* __restrict__ kvb,   // 4096 x 512
                                                          const unsigned short* __restrict__ lnb,   // 65536 x 256
                                                          unsigned short* __restrict__ gsumb) {     // 65536 x 256
  __shared__ float Pb[8][16][17];
  const int lane = threadIdx.x & 31;
  const int wv   = threadIdx.x >> 5;
  const int bh   = blockIdx.x >> 5;   // 0..63 (b, head)
  const int qc   = blockIdx.x & 31;   // 128-row q chunk
  const int b    = bh >> 2;
  const int head = bh & 3;
  const int r    = lane & 15;
  const int hi   = lane >> 4;
  const int q0   = qc * 128 + wv * 16;
  const float scale = 0.125f;

  const unsigned short* qrow = qgb + (size_t)(b * 4096 + q0 + r) * 256 + head * 64;
  v16bf a0, a1;
#pragma unroll
  for (int i = 0; i < 8; ++i) {
    a0[i]     = us2bf(qrow[hi * 8 + i]);
    a0[i + 8] = us2bf(qrow[16 + hi * 8 + i]);
    a1[i]     = us2bf(qrow[32 + hi * 8 + i]);
    a1[i + 8] = us2bf(qrow[48 + hi * 8 + i]);
  }

  float mrow[8], lrow[8];
  v8f o[4] = {};
#pragma unroll
  for (int j = 0; j < 8; ++j) {
    mrow[j] = -1e30f;
    lrow[j] = 0.f;
  }

  for (int kc = 0; kc < 16; ++kc) {
    const unsigned short* krow = kvb + (size_t)(b * 256 + kc * 16 + r) * 512 + head * 64;
    v16bf b0, b1;
#pragma unroll
    for (int i = 0; i < 16; ++i) {
      b0[i] = us2bf(krow[(i < 8 ? hi * 8 + i : 8 + hi * 8 + i)]);
      b1[i] = us2bf(krow[32 + (i < 8 ? hi * 8 + i : 8 + hi * 8 + i)]);
    }
    v8f s = {};
    s = WMMA_BF16(a0, b0, s);
    s = WMMA_BF16(a1, b1, s);

#pragma unroll
    for (int j = 0; j < 8; ++j) {
      float v  = s[j] * scale;
      float mx = v;
#pragma unroll
      for (int d = 8; d >= 1; d >>= 1) mx = fmaxf(mx, __shfl_xor(mx, d, 32));
      float mnew  = fmaxf(mrow[j], mx);
      float alpha = __expf(mrow[j] - mnew);
      float e     = __expf(v - mnew);
      float sum   = e;
#pragma unroll
      for (int d = 8; d >= 1; d >>= 1) sum += __shfl_xor(sum, d, 32);
      lrow[j] = lrow[j] * alpha + sum;
      mrow[j] = mnew;
#pragma unroll
      for (int od = 0; od < 4; ++od) o[od][j] *= alpha;
      Pb[wv][j + 8 * hi][r] = e;
    }
    __syncthreads();
    v16bf pa;
#pragma unroll
    for (int i = 0; i < 8; ++i) {
      pa[i]     = us2bf(f2bf(Pb[wv][r][hi * 8 + i]));
      pa[i + 8] = us2bf((unsigned short)0);
    }
    __syncthreads();

#pragma unroll
    for (int od = 0; od < 4; ++od) {
      v16bf vb;
#pragma unroll
      for (int i = 0; i < 16; ++i) {
        int k = (i < 8 ? hi * 8 + i : 8 + hi * 8 + i);
        unsigned short u = 0;
        if (k < 16)
          u = kvb[(size_t)(b * 256 + kc * 16 + k) * 512 + 256 + head * 64 + od * 16 + r];
        vb[i] = us2bf(u);
      }
      o[od] = WMMA_BF16(pa, vb, o[od]);
    }
  }

#pragma unroll
  for (int od = 0; od < 4; ++od)
#pragma unroll
    for (int j = 0; j < 8; ++j) {
      int m  = b * 4096 + q0 + j + 8 * hi;
      int ch = head * 64 + od * 16 + r;
      float v = o[od][j] / lrow[j];
      gsumb[(size_t)m * 256 + ch] = f2bf(v + bf2f(lnb[(size_t)m * 256 + ch]));
    }
}

// ---------------------------------------------------------------------------
// Host-side orchestration
// ---------------------------------------------------------------------------
extern "C" void kernel_launch(void* const* d_in, const int* in_sizes, int n_in,
                              void* d_out, int out_size, void* d_ws, size_t ws_size,
                              hipStream_t stream) {
  (void)in_sizes; (void)n_in; (void)out_size; (void)ws_size;

  const float* x      = (const float*)d_in[0];   // (16,4096,256)
  const float* w_qkv  = (const float*)d_in[1];   // (768,256)
  const float* b_qkv  = (const float*)d_in[2];
  const float* w_q    = (const float*)d_in[3];   // (256,256)
  const float* b_q    = (const float*)d_in[4];
  const float* w_kv   = (const float*)d_in[5];   // (512,256)
  const float* b_kv   = (const float*)d_in[6];
  const float* w_proj = (const float*)d_in[7];   // (256,256)
  const float* b_proj = (const float*)d_in[8];
  const float* gn_w   = (const float*)d_in[9];
  const float* gn_b   = (const float*)d_in[10];
  float* out          = (float*)d_out;

  const int M  = 16 * 4096;   // 65536 tokens
  const int C  = 256;

  // workspace carve
  char* ws = (char*)d_ws;
  unsigned short* xb      = (unsigned short*)ws; ws += (size_t)M * C * 2;          // x bf16
  unsigned short* qkvb    = (unsigned short*)ws; ws += (size_t)M * 768 * 2;        // qkv bf16
  float*          zbuf    = (float*)ws;          ws += (size_t)M * C * 4;          // x + grid (f32)
  unsigned short* lnb     = (unsigned short*)ws; ws += (size_t)M * C * 2;          // LN out bf16
  unsigned short* qgb     = (unsigned short*)ws; ws += (size_t)M * C * 2;          // q proj bf16
  unsigned short* pooledb = (unsigned short*)ws; ws += (size_t)4096 * C * 2;       // pooled bf16
  unsigned short* kvb     = (unsigned short*)ws; ws += (size_t)4096 * 512 * 2;     // kv bf16
  unsigned short* gsumb   = (unsigned short*)ws; ws += (size_t)M * C * 2;          // global+res bf16
  unsigned short* wqkvb   = (unsigned short*)ws; ws += (size_t)768 * C * 2;
  unsigned short* wqb     = (unsigned short*)ws; ws += (size_t)C * C * 2;
  unsigned short* wkvb    = (unsigned short*)ws; ws += (size_t)512 * C * 2;
  unsigned short* wprojb  = (unsigned short*)ws; ws += (size_t)C * C * 2;

  // 1) bf16 conversions
  f32_to_bf16_kernel<<<(M * C + 255) / 256, 256, 0, stream>>>(x, xb, M * C);
  f32_to_bf16_kernel<<<(768 * C + 255) / 256, 256, 0, stream>>>(w_qkv, wqkvb, 768 * C);
  f32_to_bf16_kernel<<<(C * C + 255) / 256, 256, 0, stream>>>(w_q, wqb, C * C);
  f32_to_bf16_kernel<<<(512 * C + 255) / 256, 256, 0, stream>>>(w_kv, wkvb, 512 * C);
  f32_to_bf16_kernel<<<(C * C + 255) / 256, 256, 0, stream>>>(w_proj, wprojb, C * C);

  // 2) qkv projection
  gemm_bias_kernel<<<dim3(M / 128, 768 / 64), 256, 0, stream>>>(xb, wqkvb, b_qkv, qkvb, nullptr, M, C, 768);

  // 3) grid-window attention + residual -> zbuf (f32)
  grid_attn_kernel<<<16384 / 4, 128, 0, stream>>>(qkvb, x, zbuf);

  // 4) channel LayerNorm
  layernorm_kernel<<<M / 8, 256, 0, stream>>>(zbuf, gn_w, gn_b, lnb);

  // 5) q projection of LN output
  gemm_bias_kernel<<<dim3(M / 128, C / 64), 256, 0, stream>>>(lnb, wqb, b_q, qgb, nullptr, M, C, C);

  // 6) 4x4 average pool + kv projection
  pool_kernel<<<(16 * 256 * 256) / 256, 256, 0, stream>>>(lnb, pooledb);
  gemm_bias_kernel<<<dim3(4096 / 128, 512 / 64), 256, 0, stream>>>(pooledb, wkvb, b_kv, kvb, nullptr, 4096, C, 512);

  // 7) pooled global attention + residual
  global_attn_kernel<<<64 * 32, 256, 0, stream>>>(qgb, kvb, lnb, gsumb);

  // 8) output projection (f32 directly into d_out, (B,N,C) order)
  gemm_bias_kernel<<<dim3(M / 128, C / 64), 256, 0, stream>>>(gsumb, wprojb, b_proj, nullptr, out, M, C, C);
}
